// ScaledDotProductAttention_46815143527073
// MI455X (gfx1250) — matrix operand order repaired and checked by reading, not compiled, not run on hardware
//
#include <hip/hip_runtime.h>
#include <hip/hip_bf16.h>

// Flash attention with causal mask + relative-position (skew-trick) bias.
// B=16, S=2048, D=64, fp32 in/out; matmuls in bf16 WMMA (f32 accumulate).
// v4b: exp2-domain online softmax (v_exp_f32 direct, scale constant folded),
//      wave-uniform split between masked (diagonal) and unmasked chunks,
//      mask realized as -inf scores only (exp2(-inf)=0, no per-element select).

#define BSZ 16
#define SEQ 2048
#define DKD 64
#define PTROWS 2080   // SEQ + 32 zero-padded relpos columns (covers c0+47 max 2079)

typedef __attribute__((ext_vector_type(16))) __bf16       v16bf;
typedef __attribute__((ext_vector_type(8)))  float        v8f;
typedef __attribute__((ext_vector_type(8)))  unsigned int v8u;
typedef __attribute__((ext_vector_type(4)))  unsigned int u32x4;

__device__ __forceinline__ unsigned short f2bfu(float f) {
  return __builtin_bit_cast(unsigned short, (__bf16)f);
}

#if defined(__HIP_DEVICE_COMPILE__) && __has_builtin(__builtin_amdgcn_exp2f)
__device__ __forceinline__ float fast_exp2(float x) { return __builtin_amdgcn_exp2f(x); }
#else
__device__ __forceinline__ float fast_exp2(float x) { return exp2f(x); }
#endif

// 16 contiguous bf16 (32B, 16B-aligned) -> WMMA operand register image.
__device__ __forceinline__ v16bf ldbf16x16(const unsigned short* p) {
  u32x4 a = *(const u32x4*)(p);
  u32x4 b = *(const u32x4*)(p + 8);
  v8u w = {a[0], a[1], a[2], a[3], b[0], b[1], b[2], b[3]};
  return __builtin_bit_cast(v16bf, w);
}

__device__ __forceinline__ void cvt4(v16bf& d, int base, float4 f) {
  d[base + 0] = (__bf16)f.x;
  d[base + 1] = (__bf16)f.y;
  d[base + 2] = (__bf16)f.z;
  d[base + 3] = (__bf16)f.w;
}

// ---------------- one-time bf16 conversion pre-passes ----------------
__global__ __launch_bounds__(256) void conv_k(const float* __restrict__ K,
                                              unsigned short* __restrict__ Kb) {
  int i = blockIdx.x * 256 + threadIdx.x;           // N = BSZ*SEQ*DKD
  Kb[i] = f2bfu(K[i]);
}
__global__ __launch_bounds__(256) void conv_vt(const float* __restrict__ V,
                                               unsigned short* __restrict__ VT) {
  int i = blockIdx.x * 256 + threadIdx.x;           // N = BSZ*SEQ*DKD
  int b = i / (SEQ * DKD), r = i % (SEQ * DKD);
  int t = r / DKD, d = r % DKD;
  VT[((size_t)b * DKD + d) * SEQ + t] = f2bfu(V[i]);  // coalesced read
}
__global__ __launch_bounds__(256) void conv_pt(const float* __restrict__ P,
                                               unsigned short* __restrict__ PT) {
  int i = blockIdx.x * 256 + threadIdx.x;           // N = PTROWS*DKD
  int col = i / DKD, d = i % DKD;
  PT[i] = (col < SEQ) ? f2bfu(P[(size_t)d * SEQ + col]) : (unsigned short)0;
}

// LDS pitches: row 16B alignment for float4 score reads; lane-halves on
// disjoint bank windows; qp scalar read stride 49 is coprime with 64 banks.
#define SC_PITCH 36   // scores 16x32 f32 (36*4 = 144 B rows, 16B aligned)
#define QP_PITCH 50   // qp band 16x48 f32
#define P_PITCH  34   // probs 16x32 bf16 (68-byte rows, 4B-aligned u32 pairs)

// log2(e)/sqrt(dk): scores go straight into exp2 domain.
#define SCALE2 (1.4426950408889634f * 0.125f)

__global__ __launch_bounds__(32)
void fa_relpos_kernel(const float* __restrict__ Q,
                      const unsigned short* __restrict__ Kb,
                      const unsigned short* __restrict__ VT,
                      const unsigned short* __restrict__ PT,
                      float* __restrict__ Out) {
  const int lane = threadIdx.x & 31;
  const int h    = lane >> 4;    // lane half
  const int nl   = lane & 15;    // N / M-within-half index
  const int b    = blockIdx.y;
  const int s0   = blockIdx.x * 16;

  __shared__ float          scLDS[16 * SC_PITCH];
  __shared__ float          qpLDS[16 * QP_PITCH];
  __shared__ unsigned short pLDS [16 * P_PITCH];
  __shared__ float          scaleLDS[16];
  __shared__ float          sumLDS[16];

  // ---- Q tile as two A operands (A-layout: k = h*16 + e) ----
  const float4* q4 = (const float4*)(Q + ((size_t)b * SEQ + (s0 + nl)) * DKD);
  v16bf qa0, qa1;
  cvt4(qa0, 0,  q4[h * 4 + 0]);  cvt4(qa0, 4,  q4[h * 4 + 1]);
  cvt4(qa0, 8,  q4[h * 4 + 2]);  cvt4(qa0, 12, q4[h * 4 + 3]);
  cvt4(qa1, 0,  q4[h * 4 + 8]);  cvt4(qa1, 4,  q4[h * 4 + 9]);
  cvt4(qa1, 8,  q4[h * 4 + 10]); cvt4(qa1, 12, q4[h * 4 + 11]);

  v8f o[4];
#pragma unroll
  for (int i = 0; i < 4; ++i) o[i] = (v8f){0.f,0.f,0.f,0.f,0.f,0.f,0.f,0.f};

  float row_max = -__builtin_inff();   // in exp2 domain
  float row_sum = 0.f;

  // One QP col-tile (16 cols starting at window col c), K=64.
  auto qp_tile = [&](int c) -> v8f {
    const unsigned short* prow = PT + (size_t)(c + nl) * DKD;
    v16bf pb0 = ldbf16x16(prow + h * 16);
    v16bf pb1 = ldbf16x16(prow + 32 + h * 16);
    v8f z = (v8f){0.f,0.f,0.f,0.f,0.f,0.f,0.f,0.f};
    v8f t = __builtin_amdgcn_wmma_f32_16x16x32_bf16(false, qa0, false, pb0,
                                                    (short)0, z, false, false);
    return __builtin_amdgcn_wmma_f32_16x16x32_bf16(false, qa1, false, pb1,
                                                   (short)0, t, false, false);
  };

  // First chunk's col-tile 0; thereafter tile 0 of chunk ch == tile 2 of ch-1.
  v8f qp_carry = qp_tile(SEQ - 16 - s0);

  const int nchunks = (s0 + 15) / 32 + 1;
  for (int ch = 0; ch < nchunks; ++ch) {
    const int t0 = ch * 32;

    // ---- QK^T: two 16x16 score tiles (B-layout: k = h*16 + e) ----
    v8f sc[2];
#pragma unroll
    for (int st = 0; st < 2; ++st) {
      const unsigned short* krow =
          Kb + ((size_t)b * SEQ + (t0 + st * 16 + nl)) * DKD;
      v16bf kb0 = ldbf16x16(krow + h * 16);
      v16bf kb1 = ldbf16x16(krow + 32 + h * 16);
      v8f z = (v8f){0.f,0.f,0.f,0.f,0.f,0.f,0.f,0.f};
      v8f t = __builtin_amdgcn_wmma_f32_16x16x32_bf16(false, qa0, false, kb0,
                                                      (short)0, z, false, false);
      sc[st] = __builtin_amdgcn_wmma_f32_16x16x32_bf16(false, qa1, false, kb1,
                                                       (short)0, t, false, false);
    }

    // ---- Relative-position band: 48 cols, tile 0 carried from prev chunk ----
    // bias[s0+i, t0+j] = QP[s0+i, c0 + 15 - i + j],  c0 = t0 - s0 + S - 16 >= 0
    const int c0 = t0 - s0 + SEQ - 16;
    v8f qp0 = qp_carry;
    v8f qp1 = qp_tile(c0 + 16);
    v8f qp2 = qp_tile(c0 + 32);
    qp_carry = qp2;

    // ---- Spill C tiles to LDS (C-layout: reg c -> M = c + 8*h, lane -> N) ----
#pragma unroll
    for (int c = 0; c < 8; ++c) {
      int row = c + 8 * h;
      scLDS[row * SC_PITCH + 0  + nl] = sc[0][c];
      scLDS[row * SC_PITCH + 16 + nl] = sc[1][c];
      qpLDS[row * QP_PITCH + 0  + nl] = qp0[c];
      qpLDS[row * QP_PITCH + 16 + nl] = qp1[c];
      qpLDS[row * QP_PITCH + 32 + nl] = qp2[c];
    }
    __syncthreads();

    // ---- Online softmax: all 32 lanes, lane = (row nl, col half h) ----
    {
      const int m = nl;
      const int jb = h * 16;                      // this lane's 16 columns
      float x[16];
      const float4* srow = (const float4*)&scLDS[m * SC_PITCH + jb];
      const float*  qrw  = &qpLDS[m * QP_PITCH + (15 - m + jb)];
      float cmax = -__builtin_inff();
      if (t0 + 31 <= s0) {
        // Fully valid chunk for every row of this tile (common path).
#pragma unroll
        for (int q = 0; q < 4; ++q) {
          float4 sv = srow[q];
#pragma unroll
          for (int r = 0; r < 4; ++r) {
            int jj = q * 4 + r;
            float scv = (r == 0) ? sv.x : (r == 1) ? sv.y : (r == 2) ? sv.z : sv.w;
            float v = (scv + qrw[jj]) * SCALE2;
            x[jj] = v;
            cmax = fmaxf(cmax, v);
          }
        }
      } else {
        // Diagonal chunk: invalid cols -> -inf (exp2(-inf)=0 later).
        const int s = s0 + m;
#pragma unroll
        for (int q = 0; q < 4; ++q) {
          float4 sv = srow[q];
#pragma unroll
          for (int r = 0; r < 4; ++r) {
            int jj = q * 4 + r;
            float scv = (r == 0) ? sv.x : (r == 1) ? sv.y : (r == 2) ? sv.z : sv.w;
            float v = (scv + qrw[jj]) * SCALE2;
            v = ((t0 + jb + jj) <= s) ? v : -__builtin_inff();
            x[jj] = v;
            cmax = fmaxf(cmax, v);
          }
        }
      }
      cmax = fmaxf(cmax, __shfl_xor(cmax, 16, 32));       // combine halves
      float nmax  = fmaxf(row_max, cmax);                 // finite: >=1 valid col
      float scale = fast_exp2(row_max - nmax);            // exp2(-inf)=0 first time
      float psum  = 0.f;
#pragma unroll
      for (int jj = 0; jj < 16; jj += 2) {
        float p0 = fast_exp2(x[jj]     - nmax);           // masked -> exactly 0
        float p1 = fast_exp2(x[jj + 1] - nmax);
        psum += p0 + p1;
        unsigned pk = (unsigned)f2bfu(p0) | ((unsigned)f2bfu(p1) << 16);
        *(unsigned*)&pLDS[m * P_PITCH + jb + jj] = pk;    // v_cvt_pk_bf16_f32
      }
      psum += __shfl_xor(psum, 16, 32);                   // combine halves
      row_sum = row_sum * scale + psum;
      row_max = nmax;
      if (h == 0) scaleLDS[m] = scale;
    }
    __syncthreads();

    // ---- Rescale accumulators by per-row factor ----
#pragma unroll
    for (int c = 0; c < 8; ++c) {
      float f = scaleLDS[c + 8 * h];
      o[0][c] *= f; o[1][c] *= f; o[2][c] *= f; o[3][c] *= f;
    }

    // ---- Rebuild probs as A operand: 8 aligned u32 (bf16 pair) LDS reads ----
    v8u pw;
#pragma unroll
    for (int j = 0; j < 8; ++j) {
      int k = h * 16 + (j << 1);   // even k
      pw[j] = *(const unsigned int*)&pLDS[nl * P_PITCH + k];
    }
    v16bf pa = __builtin_bit_cast(v16bf, pw);

    // ---- A·V: 4 output d-tiles, K = 32 keys (B from transposed bf16 V) ----
#pragma unroll
    for (int dt = 0; dt < 4; ++dt) {
      const unsigned short* vrow =
          VT + ((size_t)b * DKD + dt * 16 + nl) * SEQ + t0;
      v16bf vb = ldbf16x16(vrow + h * 16);
      o[dt] = __builtin_amdgcn_wmma_f32_16x16x32_bf16(false, pa, false, vb,
                                                      (short)0, o[dt], false, false);
    }
    __syncthreads();   // fence LDS before next chunk overwrites
  }

  // ---- Normalize by row sums and store ----
  if (lane < 16) sumLDS[lane] = row_sum;   // halves hold identical sums
  __syncthreads();
#pragma unroll
  for (int c = 0; c < 8; ++c) {
    int row = c + 8 * h;
    float inv = 1.0f / sumLDS[row];
    float* orow = Out + ((size_t)b * SEQ + (s0 + row)) * DKD + nl;
    orow[0]  = o[0][c] * inv;
    orow[16] = o[1][c] * inv;
    orow[32] = o[2][c] * inv;
    orow[48] = o[3][c] * inv;
  }
}

extern "C" void kernel_launch(void* const* d_in, const int* in_sizes, int n_in,
                              void* d_out, int out_size, void* d_ws, size_t ws_size,
                              hipStream_t stream) {
  const float* Q  = (const float*)d_in[0];
  const float* K  = (const float*)d_in[1];
  const float* V  = (const float*)d_in[2];
  const float* Kp = (const float*)d_in[3];
  float* Out      = (float*)d_out;

  // Workspace layout (bf16): Kb [B][S][D] | VT [B][D][S] | PT [PTROWS][D]
  unsigned short* wsKb = (unsigned short*)d_ws;
  unsigned short* wsVT = wsKb + (size_t)BSZ * SEQ * DKD;
  unsigned short* wsPT = wsVT + (size_t)BSZ * SEQ * DKD;   // total ~8.7 MB

  const int NKV = BSZ * SEQ * DKD;                 // 2,097,152
  conv_k <<<NKV / 256, 256, 0, stream>>>(K, wsKb);
  conv_vt<<<NKV / 256, 256, 0, stream>>>(V, wsVT);
  conv_pt<<<(PTROWS * DKD) / 256, 256, 0, stream>>>(Kp, wsPT);

  dim3 grid(SEQ / 16, BSZ);   // 128 query tiles x 16 batches
  dim3 block(32);             // one wave32 per query tile
  fa_relpos_kernel<<<grid, block, 0, stream>>>(Q, wsKb, wsVT, wsPT, Out);
}
